// GPT_26010321945295
// MI455X (gfx1250) — matrix operand order repaired and checked by reading, not compiled, not run on hardware
//
#include <hip/hip_runtime.h>
#include <hip/hip_bf16.h>
#include <math.h>

// ---------------------------------------------------------------------------
// Model dims (match reference)
// ---------------------------------------------------------------------------
#define D_    1024
#define NH_   16
#define HD_   64
#define HID_  4096
#define E_    8
#define V_    50304
#define T_    1024
#define B_    4
#define REC_  2
#define ROWS_ (B_*T_)          // 4096 token rows
#define ATT_SCALE_ 0.125f      // 1/sqrt(64)

typedef __bf16 bf16;
typedef __attribute__((ext_vector_type(16))) __bf16 bf16x16;
typedef __attribute__((ext_vector_type(8)))  __bf16 bf16x8;
typedef __attribute__((ext_vector_type(8)))  float  f32x8;
typedef __attribute__((ext_vector_type(4)))  unsigned int tdm_u32x4;
typedef __attribute__((ext_vector_type(4)))  int          tdm_i32x4;
typedef __attribute__((ext_vector_type(8)))  int          tdm_i32x8;

// Tensor Data Mover availability (device pass only)
#if defined(__gfx1250__) && __has_builtin(__builtin_amdgcn_tensor_load_to_lds)
#define GEMM_TDM 1
#warning "CDNA5: TDM tensor_load_to_lds path ENABLED for dense GEMM A-tiles"
#else
#define GEMM_TDM 0
#warning "CDNA5: TDM builtin not available on this pass; async-to-LDS fallback"
#endif

// Build a 16-element bf16 fragment from two 16B LDS chunks at p and p+off.
__device__ inline bf16x16 ld_frag(const bf16* p, int off_elems)
{
    bf16x8 lo = *(const bf16x8*)p;
    bf16x8 hi = *(const bf16x8*)(p + off_elems);
    bf16x16 r;
    #pragma unroll
    for (int t = 0; t < 8; ++t) { r[t] = lo[t]; r[t + 8] = hi[t]; }
    return r;
}

// ---------------------------------------------------------------------------
// Embedding: x[row,:] = wte[idx[row],:] + wpe[row%T,:]
// ---------------------------------------------------------------------------
__global__ __launch_bounds__(256)
void embed_kernel(const int* __restrict__ idx, const float* __restrict__ wte,
                  const float* __restrict__ wpe, float* __restrict__ x)
{
    int row = blockIdx.x;
    int c   = threadIdx.x * 4;
    int tok = idx[row];
    int t   = row & (T_ - 1);
    float4 e = *(const float4*)(wte + (size_t)tok * D_ + c);
    float4 p = *(const float4*)(wpe + (size_t)t   * D_ + c);
    e.x += p.x; e.y += p.y; e.z += p.z; e.w += p.w;
    *(float4*)(x + (size_t)row * D_ + c) = e;
}

// ---------------------------------------------------------------------------
// LayerNorm over D=1024. One block per row; optional bf16 / f32 outputs.
// Input row = row_off + blockIdx.x*row_step, output row = blockIdx.x.
// ---------------------------------------------------------------------------
__global__ __launch_bounds__(256)
void ln_kernel(const float* __restrict__ x, const float* __restrict__ g,
               const float* __restrict__ b, bf16* __restrict__ outb,
               float* __restrict__ outf, int row_off, int row_step)
{
    int in_row = row_off + blockIdx.x * row_step;
    const float* xr = x + (size_t)in_row * D_;
    int tid  = threadIdx.x;
    int lane = tid & 31;
    int wave = tid >> 5;

    float4 v = *(const float4*)(xr + tid * 4);
    float s  = v.x + v.y + v.z + v.w;
    float s2 = v.x*v.x + v.y*v.y + v.z*v.z + v.w*v.w;
    #pragma unroll
    for (int o = 16; o; o >>= 1) { s += __shfl_xor(s, o); s2 += __shfl_xor(s2, o); }

    __shared__ float rs[8], rs2[8];
    if (lane == 0) { rs[wave] = s; rs2[wave] = s2; }
    __syncthreads();
    if (tid == 0) {
        float a = 0.f, a2 = 0.f;
        for (int i = 0; i < 8; ++i) { a += rs[i]; a2 += rs2[i]; }
        rs[0] = a; rs2[0] = a2;
    }
    __syncthreads();
    float mean = rs[0]  * (1.f / D_);
    float var  = rs2[0] * (1.f / D_) - mean * mean;
    float rstd = rsqrtf(var + 1e-5f);

    int c = tid * 4;
    float4 gw = *(const float4*)(g + c);
    float4 bw = *(const float4*)(b + c);
    float y0 = (v.x - mean) * rstd * gw.x + bw.x;
    float y1 = (v.y - mean) * rstd * gw.y + bw.y;
    float y2 = (v.z - mean) * rstd * gw.z + bw.z;
    float y3 = (v.w - mean) * rstd * gw.w + bw.w;
    size_t obase = (size_t)blockIdx.x * D_ + c;
    if (outb) {
        bf16* op = outb + obase;
        op[0] = (bf16)y0; op[1] = (bf16)y1; op[2] = (bf16)y2; op[3] = (bf16)y3;
    }
    if (outf) {
        float4 o = { y0, y1, y2, y3 };
        *(float4*)(outf + obase) = o;
    }
}

// ---------------------------------------------------------------------------
// WMMA GEMM with optional gathered-A rows (GIN) and scattered/gated output
// rows (GOUT) for sparse MoE dispatch. 128x128 tile, K-step 32, double-
// buffered LDS; dense A-tiles via TDM (or async-to-LDS), gathered A-tiles via
// per-lane async-to-LDS. W fp32->bf16 converted on the way into LDS.
// ---------------------------------------------------------------------------
#define BM 128
#define BN 128
#define BK 32
#define LDSK 40   // padded bf16 row stride: 80B rows (pad = 4 DWORDs per 16)

template<bool ACC, bool SCALEROW, bool GIN, bool GOUT>
__global__ __launch_bounds__(256)
void gemm_kernel(const bf16* __restrict__ A, const float* __restrict__ W,
                 const float* __restrict__ bias, float* __restrict__ C,
                 const float* __restrict__ scale, int sstride,
                 const int* __restrict__ rowmap, const int* __restrict__ countp,
                 int M, int N, int K)
{
    __shared__ __align__(16) bf16 As[2][BM * LDSK];
    __shared__ __align__(16) bf16 Bs[2][BN * LDSK];

    const int tid  = threadIdx.x;
    const int lane = tid & 31;
    const int wave = tid >> 5;
    const int m0   = blockIdx.y * BM;
    const int n0   = blockIdx.x * BN;
    const int wm0  = (wave & 1) * 64;
    const int wn0  = (wave >> 1) * 32;

    int cnt = M;
    if constexpr (GIN || GOUT) {
        cnt = countp[0];
        if (m0 >= cnt) return;      // block-uniform early exit
    }

    const int trow  = tid >> 1;        // 0..127: tile row handled by this thread
    const int thalf = (tid & 1) * 16;  // element offset within 32-wide K slab

    int arow_g;                        // global A row for this thread's tile row
    if constexpr (GIN) arow_g = rowmap[m0 + trow];
    else               arow_g = m0 + trow;

    f32x8 acc[4][2];
    #pragma unroll
    for (int i = 0; i < 4; ++i)
        #pragma unroll
        for (int j = 0; j < 2; ++j)
            #pragma unroll
            for (int k = 0; k < 8; ++k) acc[i][j][k] = 0.f;

    // ---- stage A tile (bf16 128x32) into LDS buffer `buf` ----
    auto stage_A = [&](int kk, int buf) {
#if GEMM_TDM
        if constexpr (!GIN) {
            if (tid == 0) {
                const unsigned long long ga =
                    (unsigned long long)(const void*)(A + (size_t)m0 * K + kk);
                const unsigned ldsa = (unsigned)(unsigned long long)(void*)&As[buf][0];
                tdm_u32x4 g0 = { 1u, ldsa, (unsigned)ga,
                                 (unsigned)(ga >> 32) | (2u << 30) };
                tdm_i32x8 g1 = {
                    (int)((1u << 16) | (1u << 20) | (3u << 22) | (3u << 25)),
                    (int)(((unsigned)K & 0xFFFFu) << 16),
                    (int)(((unsigned)K >> 16) | (((unsigned)M & 0xFFFFu) << 16)),
                    (int)(((unsigned)M >> 16) | (32u << 16)),
                    (int)128,
                    (int)K,
                    0, 0 };
                tdm_i32x4 z4 = { 0, 0, 0, 0 };
#if __clang_major__ >= 23
                tdm_i32x8 z8 = { 0, 0, 0, 0, 0, 0, 0, 0 };
                __builtin_amdgcn_tensor_load_to_lds(g0, g1, z4, z4, z8, 0);
#else
                __builtin_amdgcn_tensor_load_to_lds(g0, g1, z4, z4, 0);
#endif
            }
        } else {
            const bf16* ag = A + (size_t)arow_g * K + kk + thalf;
            bf16*       al = &As[buf][trow * LDSK + thalf];
            unsigned           lds_off = (unsigned)(unsigned long long)(void*)al;
            unsigned long long gaddr   = (unsigned long long)ag;
            asm volatile("global_load_async_to_lds_b128 %0, %1, off"
                         :: "v"(lds_off), "v"(gaddr) : "memory");
            asm volatile("global_load_async_to_lds_b128 %0, %1, off offset:16"
                         :: "v"(lds_off), "v"(gaddr) : "memory");
        }
#elif defined(__gfx1250__)
        const bf16* ag = A + (size_t)arow_g * K + kk + thalf;
        bf16*       al = &As[buf][trow * LDSK + thalf];
        unsigned           lds_off = (unsigned)(unsigned long long)(void*)al;
        unsigned long long gaddr   = (unsigned long long)ag;
        asm volatile("global_load_async_to_lds_b128 %0, %1, off"
                     :: "v"(lds_off), "v"(gaddr) : "memory");
        asm volatile("global_load_async_to_lds_b128 %0, %1, off offset:16"
                     :: "v"(lds_off), "v"(gaddr) : "memory");
#else
        const bf16* ag = A + (size_t)arow_g * K + kk + thalf;
        bf16*       al = &As[buf][trow * LDSK + thalf];
        *(bf16x8*)(al)     = *(const bf16x8*)(ag);
        *(bf16x8*)(al + 8) = *(const bf16x8*)(ag + 8);
#endif
    };
    auto wait_stage = [&]() {
#if GEMM_TDM
        if constexpr (GIN) { asm volatile("s_wait_asynccnt 0" ::: "memory"); }
        else               { __builtin_amdgcn_s_wait_tensorcnt(0); }
#elif defined(__gfx1250__)
        asm volatile("s_wait_asynccnt 0" ::: "memory");
#endif
    };
    auto load_B = [&](int kk, float4* w) {
        const float* wg = W + (size_t)(n0 + trow) * K + kk + thalf;
        w[0] = ((const float4*)wg)[0];
        w[1] = ((const float4*)wg)[1];
        w[2] = ((const float4*)wg)[2];
        w[3] = ((const float4*)wg)[3];
        __builtin_prefetch(wg + BK, 0, 1);
    };
    auto store_B = [&](const float4* w, int buf) {
        bf16* bl = &Bs[buf][trow * LDSK + thalf];
        bf16x8 p0 = { (bf16)w[0].x, (bf16)w[0].y, (bf16)w[0].z, (bf16)w[0].w,
                      (bf16)w[1].x, (bf16)w[1].y, (bf16)w[1].z, (bf16)w[1].w };
        bf16x8 p1 = { (bf16)w[2].x, (bf16)w[2].y, (bf16)w[2].z, (bf16)w[2].w,
                      (bf16)w[3].x, (bf16)w[3].y, (bf16)w[3].z, (bf16)w[3].w };
        *(bf16x8*)(bl)     = p0;
        *(bf16x8*)(bl + 8) = p1;
    };

    const int nkt = K / BK;
    float4 wreg[4];
    stage_A(0, 0);
    load_B(0, wreg);
    store_B(wreg, 0);
    wait_stage();

    for (int kt = 0; kt < nkt; ++kt) {
        const int cur = kt & 1, nxt = cur ^ 1;
        __syncthreads();   // stage `cur` fully written; buf `nxt` fully consumed

        if (kt + 1 < nkt) {
            stage_A((kt + 1) * BK, nxt);   // overlap with compute below
            load_B((kt + 1) * BK, wreg);
        }

        const int khalf  = (lane < 16) ? 0 : 8;
        const int kbaseB = (lane < 16) ? 0 : 16;
        bf16x16 af[4];
        #pragma unroll
        for (int i = 0; i < 4; ++i)
            af[i] = ld_frag(&As[cur][(wm0 + 16*i + (lane & 15)) * LDSK + khalf], 16);
        bf16x16 bfg[2];
        #pragma unroll
        for (int j = 0; j < 2; ++j)
            bfg[j] = ld_frag(&Bs[cur][(wn0 + 16*j + (lane & 15)) * LDSK + khalf], 16);

        #pragma unroll
        for (int i = 0; i < 4; ++i)
            #pragma unroll
            for (int j = 0; j < 2; ++j)
                acc[i][j] = __builtin_amdgcn_wmma_f32_16x16x32_bf16(
                    false, af[i], false, bfg[j], (short)0, acc[i][j], false, false);

        if (kt + 1 < nkt) store_B(wreg, nxt);
        wait_stage();
    }

    // ---- epilogue ----
    const int colb = n0 + wn0 + (lane & 15);
    const int rowb = m0 + wm0 + ((lane >> 4) << 3);
    #pragma unroll
    for (int j = 0; j < 2; ++j) {
        const int col = colb + 16 * j;
        const float bv = bias ? bias[col] : 0.f;
        #pragma unroll
        for (int i = 0; i < 4; ++i) {
            #pragma unroll
            for (int r = 0; r < 8; ++r) {
                const int grow = rowb + 16 * i + r;
                if ((GIN || GOUT) && grow >= cnt) continue;   // pad rows
                const int orow = GOUT ? rowmap[grow] : grow;
                float v = acc[i][j][r] + bv;
                if (SCALEROW) v *= scale[(size_t)orow * sstride];
                const size_t off = (size_t)orow * N + col;
                if (ACC) v += C[off];
                C[off] = v;
            }
        }
    }
}

// ---------------------------------------------------------------------------
// WMMA flash attention (unchanged from round 2).
// ---------------------------------------------------------------------------
#define FA_QB 128
#define FA_KB 32
#define QSTR  72
#define VSTR  40
#define PSTR  40

__global__ __launch_bounds__(256)
void fattn_kernel(const float* __restrict__ qkv, bf16* __restrict__ y)
{
    __shared__ __align__(16) bf16 Qs[FA_QB * QSTR];
    __shared__ __align__(16) bf16 Ks[FA_KB * QSTR];
    __shared__ __align__(16) bf16 VTs[HD_ * VSTR];
    __shared__ __align__(16) bf16 Ps[8 * 16 * PSTR];

    const int tid  = threadIdx.x;
    const int lane = tid & 31;
    const int wave = tid >> 5;
    const int q0   = blockIdx.x * FA_QB;
    const int bh   = blockIdx.y;
    const int bb   = bh >> 4;
    const int h    = bh & 15;

    const float* qbase = qkv + (size_t)bb * T_ * 3 * D_ + h * HD_;
    const float* kbase = qbase + D_;
    const float* vbase = qbase + 2 * D_;

    {
        const int r  = tid >> 1;
        const int c0 = (tid & 1) * 32;
        const float* qp = qbase + (size_t)(q0 + r) * 3 * D_ + c0;
        bf16* dst = &Qs[r * QSTR + c0];
        #pragma unroll
        for (int c = 0; c < 32; c += 8) {
            float4 a = *(const float4*)(qp + c);
            float4 d = *(const float4*)(qp + c + 4);
            bf16x8 pk = { (bf16)(a.x*ATT_SCALE_), (bf16)(a.y*ATT_SCALE_),
                          (bf16)(a.z*ATT_SCALE_), (bf16)(a.w*ATT_SCALE_),
                          (bf16)(d.x*ATT_SCALE_), (bf16)(d.y*ATT_SCALE_),
                          (bf16)(d.z*ATT_SCALE_), (bf16)(d.w*ATT_SCALE_) };
            *(bf16x8*)(dst + c) = pk;
        }
    }
    __syncthreads();

    const int khalf = (lane < 16) ? 0 : 8;
    const int kb16  = (lane < 16) ? 0 : 16;
    const int hi8   = (lane >> 4) << 3;
    bf16x16 qf[2];
    #pragma unroll
    for (int ks = 0; ks < 2; ++ks)
        qf[ks] = ld_frag(&Qs[(wave*16 + (lane & 15)) * QSTR + 32*ks + khalf], 16);

    f32x8 o[4];
    #pragma unroll
    for (int dt = 0; dt < 4; ++dt)
        #pragma unroll
        for (int k = 0; k < 8; ++k) o[dt][k] = 0.f;
    float m8[8], l8[8];
    #pragma unroll
    for (int r = 0; r < 8; ++r) { m8[r] = -1e30f; l8[r] = 0.f; }

    const int myqmax = q0 + wave * 16 + 15;
    for (int kb = 0; kb < q0 + FA_QB; kb += FA_KB) {
        __syncthreads();
        {
            const int kr = tid >> 3;
            const int c0 = (tid & 7) * 8;
            const float* kp = kbase + (size_t)(kb + kr) * 3 * D_ + c0;
            float4 a = *(const float4*)kp;
            float4 d = *(const float4*)(kp + 4);
            bf16x8 pk = { (bf16)a.x, (bf16)a.y, (bf16)a.z, (bf16)a.w,
                          (bf16)d.x, (bf16)d.y, (bf16)d.z, (bf16)d.w };
            *(bf16x8*)(&Ks[kr * QSTR + c0]) = pk;
            const float* vp = vbase + (size_t)(kb + kr) * 3 * D_ + c0;
            float4 va = *(const float4*)vp;
            float4 vd = *(const float4*)(vp + 4);
            bf16 tv[8] = { (bf16)va.x, (bf16)va.y, (bf16)va.z, (bf16)va.w,
                           (bf16)vd.x, (bf16)vd.y, (bf16)vd.z, (bf16)vd.w };
            #pragma unroll
            for (int d2 = 0; d2 < 8; ++d2) VTs[(c0 + d2) * VSTR + kr] = tv[d2];
        }
        __syncthreads();

        if (kb <= myqmax) {
            f32x8 s[2];
            #pragma unroll
            for (int jj = 0; jj < 2; ++jj) {
                #pragma unroll
                for (int k = 0; k < 8; ++k) s[jj][k] = 0.f;
                #pragma unroll
                for (int ks = 0; ks < 2; ++ks) {
                    bf16x16 kf = ld_frag(
                        &Ks[(16*jj + (lane & 15)) * QSTR + 32*ks + khalf], 16);
                    s[jj] = __builtin_amdgcn_wmma_f32_16x16x32_bf16(
                        false, qf[ks], false, kf, (short)0, s[jj], false, false);
                }
            }
            const int rbase = q0 + wave * 16 + hi8;
            const int key0  = kb + (lane & 15);
            #pragma unroll
            for (int r = 0; r < 8; ++r) {
                const int qrow = rbase + r;
                if (key0      > qrow) s[0][r] = -1e30f;
                if (key0 + 16 > qrow) s[1][r] = -1e30f;
                float mv = fmaxf(s[0][r], s[1][r]);
                #pragma unroll
                for (int o2 = 8; o2; o2 >>= 1) mv = fmaxf(mv, __shfl_xor(mv, o2));
                const float mn = fmaxf(m8[r], mv);
                const float al = __expf(m8[r] - mn);
                const float e0 = __expf(s[0][r] - mn);
                const float e1 = __expf(s[1][r] - mn);
                s[0][r] = e0; s[1][r] = e1;
                float rsum = e0 + e1;
                #pragma unroll
                for (int o2 = 8; o2; o2 >>= 1) rsum += __shfl_xor(rsum, o2);
                l8[r] = l8[r] * al + rsum;
                m8[r] = mn;
                o[0][r] *= al; o[1][r] *= al; o[2][r] *= al; o[3][r] *= al;
            }
            bf16* pw = &Ps[wave * 16 * PSTR];
            #pragma unroll
            for (int r = 0; r < 8; ++r) {
                const int row = hi8 + r;
                pw[row * PSTR + (lane & 15)]      = (bf16)s[0][r];
                pw[row * PSTR + 16 + (lane & 15)] = (bf16)s[1][r];
            }
            bf16x16 pf = ld_frag(&pw[(lane & 15) * PSTR + khalf], 16);
            #pragma unroll
            for (int dt = 0; dt < 4; ++dt) {
                bf16x16 vf = ld_frag(&VTs[(16*dt + (lane & 15)) * VSTR + khalf], 16);
                o[dt] = __builtin_amdgcn_wmma_f32_16x16x32_bf16(
                    false, pf, false, vf, (short)0, o[dt], false, false);
            }
        }
    }

    #pragma unroll
    for (int dt = 0; dt < 4; ++dt) {
        const int col = h * HD_ + 16 * dt + (lane & 15);
        #pragma unroll
        for (int r = 0; r < 8; ++r) {
            const int row = q0 + wave * 16 + hi8 + r;
            y[(size_t)(bb * T_ + row) * D_ + col] = (bf16)(o[dt][r] / l8[r]);
        }
    }
}

// ---------------------------------------------------------------------------
// SwiGLU: h = silu(g) * u; optionally clamp to live token rows.
// ---------------------------------------------------------------------------
__global__ __launch_bounds__(256)
void silu_mul_kernel(const float* __restrict__ g, const float* __restrict__ u,
                     bf16* __restrict__ h, int n, const int* __restrict__ countp)
{
    if (countp) { int ne = countp[0] * HID_; n = ne < n ? ne : n; }
    int i = blockIdx.x * blockDim.x + threadIdx.x;
    if (i < n) {
        float gv = g[i];
        float hv = gv / (1.f + __expf(-gv)) * u[i];
        h[i] = (bf16)hv;
    }
}

// ---------------------------------------------------------------------------
// Noisy top-2 router: gates + top-2 expert indices per token.
// ---------------------------------------------------------------------------
__device__ inline float softplusf(float x)
{
    return fmaxf(x, 0.f) + log1pf(__expf(-fabsf(x)));
}

__global__ __launch_bounds__(256)
void router_kernel(const bf16* __restrict__ xn,
                   const float* __restrict__ rw, const float* __restrict__ rb,
                   const float* __restrict__ nw, const float* __restrict__ nb,
                   const float* __restrict__ noise, float* __restrict__ gate,
                   int* __restrict__ topk)
{
    const int tokid = (blockIdx.x * blockDim.x + threadIdx.x) >> 5;
    const int lane  = threadIdx.x & 31;

    const bf16* xr = xn + (size_t)tokid * D_;
    float xl[32];
    #pragma unroll
    for (int i = 0; i < 32; ++i) xl[i] = (float)xr[lane + 32 * i];

    float noisy[8];
    #pragma unroll
    for (int e = 0; e < 8; ++e) {
        const float* rp = rw + (size_t)e * D_;
        const float* np = nw + (size_t)e * D_;
        float s1 = 0.f, s2 = 0.f;
        #pragma unroll
        for (int i = 0; i < 32; ++i) {
            const float xv = xl[i];
            s1 += xv * rp[lane + 32 * i];
            s2 += xv * np[lane + 32 * i];
        }
        #pragma unroll
        for (int o = 16; o; o >>= 1) { s1 += __shfl_xor(s1, o); s2 += __shfl_xor(s2, o); }
        noisy[e] = (s1 + rb[e]) + noise[(size_t)tokid * E_ + e] * softplusf(s2 + nb[e]);
    }

    int i1 = -1, i2 = -1;
    float v1 = -1e30f, v2 = -1e30f;
    #pragma unroll
    for (int e = 0; e < 8; ++e) {
        const float v = noisy[e];
        if (v > v1)      { v2 = v1; i2 = i1; v1 = v; i1 = e; }
        else if (v > v2) { v2 = v; i2 = e; }
    }
    const float e2 = __expf(v2 - v1);
    const float inv = 1.f / (1.f + e2);
    if (lane < 8)
        gate[(size_t)tokid * E_ + lane] =
            (lane == i1) ? inv : ((lane == i2) ? e2 * inv : 0.f);
    if (lane == 0) {
        topk[(size_t)tokid * 2]     = i1;
        topk[(size_t)tokid * 2 + 1] = i2;
    }
}

// ---------------------------------------------------------------------------
// Deterministic per-expert token compaction (no atomics): one block per
// expert; ballot + popcount prefix within waves, LDS scan across waves.
// Pads list to a BM multiple with token 0 and writes the live count.
// ---------------------------------------------------------------------------
__global__ __launch_bounds__(256)
void compact_kernel(const int* __restrict__ topk, int* __restrict__ lists,
                    int* __restrict__ counts)
{
    const int e    = blockIdx.x;
    const int tid  = threadIdx.x;
    const int lane = tid & 31;
    const int wave = tid >> 5;
    __shared__ int wtot[8];
    __shared__ int basecnt;
    if (tid == 0) basecnt = 0;
    __syncthreads();

    int* list = lists + (size_t)e * ROWS_;
    for (int t0 = 0; t0 < ROWS_; t0 += 256) {
        const int tok = t0 + tid;
        const bool sel = (topk[(size_t)tok * 2] == e) || (topk[(size_t)tok * 2 + 1] == e);
        const unsigned m32 = (unsigned)__ballot(sel);
        if (lane == 0) wtot[wave] = __popc(m32);
        __syncthreads();
        int wbase = basecnt;
        for (int w = 0; w < wave; ++w) wbase += wtot[w];
        const int prefix = __popc(m32 & ((1u << lane) - 1u));
        if (sel) list[wbase + prefix] = tok;
        int tot = 0;
        for (int w = 0; w < 8; ++w) tot += wtot[w];
        __syncthreads();
        if (tid == 0) basecnt += tot;
        __syncthreads();
    }
    const int cnt    = basecnt;
    const int padded = (cnt + BM - 1) & ~(BM - 1);
    for (int i = cnt + tid; i < padded; i += 256) list[i] = 0;
    if (tid == 0) counts[e] = cnt;
}

// ---------------------------------------------------------------------------
// LM head on the 4 last-token rows: one block per vocab row.
// ---------------------------------------------------------------------------
__global__ __launch_bounds__(256)
void lmhead_kernel(const float* __restrict__ xnf, const float* __restrict__ wte,
                   float* __restrict__ out)
{
    const int v    = blockIdx.x;
    const int tid  = threadIdx.x;
    const int lane = tid & 31;
    const int wave = tid >> 5;

    float acc[4] = {0.f, 0.f, 0.f, 0.f};
    const float* wp = wte + (size_t)v * D_;
    for (int d = tid; d < D_; d += 256) {
        const float wv = wp[d];
        acc[0] += wv * xnf[d];
        acc[1] += wv * xnf[D_ + d];
        acc[2] += wv * xnf[2 * D_ + d];
        acc[3] += wv * xnf[3 * D_ + d];
    }
    #pragma unroll
    for (int o = 16; o; o >>= 1) {
        acc[0] += __shfl_xor(acc[0], o);
        acc[1] += __shfl_xor(acc[1], o);
        acc[2] += __shfl_xor(acc[2], o);
        acc[3] += __shfl_xor(acc[3], o);
    }
    __shared__ float sm[8][4];
    if (lane == 0) { sm[wave][0] = acc[0]; sm[wave][1] = acc[1];
                     sm[wave][2] = acc[2]; sm[wave][3] = acc[3]; }
    __syncthreads();
    if (tid < 4) {
        float s = 0.f;
        for (int w = 0; w < 8; ++w) s += sm[w][tid];
        out[(size_t)tid * V_ + v] = s;
    }
}

// ---------------------------------------------------------------------------
// Host-side GEMM dispatchers
// ---------------------------------------------------------------------------
static inline void gemm_dense(hipStream_t s, const bf16* A, const float* W,
                              const float* bias, float* C, int M, int N, int K,
                              bool accumulate)
{
    dim3 grid(N / BN, M / BM), blk(256);
    if (accumulate)
        gemm_kernel<true, false, false, false><<<grid, blk, 0, s>>>(
            A, W, bias, C, nullptr, 0, nullptr, nullptr, M, N, K);
    else
        gemm_kernel<false, false, false, false><<<grid, blk, 0, s>>>(
            A, W, bias, C, nullptr, 0, nullptr, nullptr, M, N, K);
}
static inline void gemm_gather(hipStream_t s, const bf16* A, const float* W,
                               const float* bias, float* C,
                               const int* rowmap, const int* countp, int N, int K)
{
    dim3 grid(N / BN, ROWS_ / BM), blk(256);
    gemm_kernel<false, false, true, false><<<grid, blk, 0, s>>>(
        A, W, bias, C, nullptr, 0, rowmap, countp, ROWS_, N, K);
}
static inline void gemm_scatter(hipStream_t s, const bf16* A, const float* W,
                                const float* bias, float* C,
                                const float* scale, int sstride,
                                const int* rowmap, const int* countp, int N, int K)
{
    dim3 grid(N / BN, ROWS_ / BM), blk(256);
    gemm_kernel<true, true, false, true><<<grid, blk, 0, s>>>(
        A, W, bias, C, scale, sstride, rowmap, countp, ROWS_, N, K);
}

// ---------------------------------------------------------------------------
// kernel_launch
// ---------------------------------------------------------------------------
extern "C" void kernel_launch(void* const* d_in, const int* in_sizes, int n_in,
                              void* d_out, int out_size, void* d_ws, size_t ws_size,
                              hipStream_t stream)
{
    (void)in_sizes; (void)n_in; (void)out_size; (void)ws_size;

    const float* wte    = (const float*)d_in[0];
    const float* wpe    = (const float*)d_in[1];
    const float* b0ln1w = (const float*)d_in[2];
    const float* b0ln1b = (const float*)d_in[3];
    const float* b0aw   = (const float*)d_in[4];
    const float* b0ab   = (const float*)d_in[5];
    const float* b0pw   = (const float*)d_in[6];
    const float* b0pb   = (const float*)d_in[7];
    const float* b0ln2w = (const float*)d_in[8];
    const float* b0ln2b = (const float*)d_in[9];
    const float* b0gw   = (const float*)d_in[10];
    const float* b0gb   = (const float*)d_in[11];
    const float* b0uw   = (const float*)d_in[12];
    const float* b0ub   = (const float*)d_in[13];
    const float* b0dw   = (const float*)d_in[14];
    const float* b0db   = (const float*)d_in[15];
    const float* b1ln1w = (const float*)d_in[16];
    const float* b1ln1b = (const float*)d_in[17];
    const float* b1aw   = (const float*)d_in[18];
    const float* b1ab   = (const float*)d_in[19];
    const float* b1pw   = (const float*)d_in[20];
    const float* b1pb   = (const float*)d_in[21];
    const float* b1ln2w = (const float*)d_in[22];
    const float* b1ln2b = (const float*)d_in[23];
    const float* rw     = (const float*)d_in[24];
    const float* rb     = (const float*)d_in[25];
    const float* nw     = (const float*)d_in[26];
    const float* nb     = (const float*)d_in[27];
    const float* sgw    = (const float*)d_in[28];
    const float* sgb    = (const float*)d_in[29];
    const float* suw    = (const float*)d_in[30];
    const float* sub_   = (const float*)d_in[31];
    const float* sdw    = (const float*)d_in[32];
    const float* sdb    = (const float*)d_in[33];
    const float* egw    = (const float*)d_in[34];
    const float* egb    = (const float*)d_in[35];
    const float* euw    = (const float*)d_in[36];
    const float* eub    = (const float*)d_in[37];
    const float* edw    = (const float*)d_in[38];
    const float* edb    = (const float*)d_in[39];
    const float* lnfw   = (const float*)d_in[40];
    const float* lnfb   = (const float*)d_in[41];
    const int*   idx    = (const int*)d_in[42];
    const float* rnois  = (const float*)d_in[43];

    char*  base = (char*)d_ws;
    size_t off  = 0;
    auto take = [&](size_t bytes) -> void* {
        void* p = base + off;
        off += (bytes + 255) & ~(size_t)255;
        return p;
    };
    float* x     = (float*)take((size_t)ROWS_ * D_ * 4);
    bf16*  xn    = (bf16*) take((size_t)ROWS_ * D_ * 2);
    float* qkv   = (float*)take((size_t)ROWS_ * 3 * D_ * 4);
    bf16*  ybf   = (bf16*) take((size_t)ROWS_ * D_ * 2);
    float* gbuf  = (float*)take((size_t)ROWS_ * HID_ * 4);
    float* ubuf  = (float*)take((size_t)ROWS_ * HID_ * 4);
    bf16*  hbf   = (bf16*) take((size_t)ROWS_ * HID_ * 2);
    float* gate  = (float*)take((size_t)ROWS_ * E_ * 4);
    int*   topk  = (int*)  take((size_t)ROWS_ * 2 * 4);
    int*   lists = (int*)  take((size_t)E_ * ROWS_ * 4);
    int*   cnts  = (int*)  take((size_t)E_ * 4);
    float* xnf   = (float*)take((size_t)B_ * D_ * 4);

    auto run_attn = [&](const float* ln1w, const float* ln1b, const float* aw,
                        const float* ab, const float* pw, const float* pb) {
        ln_kernel<<<ROWS_, 256, 0, stream>>>(x, ln1w, ln1b, xn, nullptr, 0, 1);
        gemm_dense(stream, xn, aw, ab, qkv, ROWS_, 3 * D_, D_, false);
        fattn_kernel<<<dim3(T_ / FA_QB, B_ * NH_), 256, 0, stream>>>(qkv, ybf);
        gemm_dense(stream, ybf, pw, pb, x, ROWS_, D_, D_, true);
    };
    auto run_dense_mlp = [&](const float* gw, const float* gb, const float* uw,
                             const float* ub, const float* dw, const float* db) {
        gemm_dense(stream, xn, gw, gb, gbuf, ROWS_, HID_, D_, false);
        gemm_dense(stream, xn, uw, ub, ubuf, ROWS_, HID_, D_, false);
        silu_mul_kernel<<<(ROWS_ * HID_) / 256, 256, 0, stream>>>(
            gbuf, ubuf, hbf, ROWS_ * HID_, nullptr);
        gemm_dense(stream, hbf, dw, db, x, ROWS_, D_, HID_, true);
    };

    embed_kernel<<<ROWS_, 256, 0, stream>>>(idx, wte, wpe, x);

    for (int r = 0; r < REC_; ++r) {
        run_attn(b0ln1w, b0ln1b, b0aw, b0ab, b0pw, b0pb);
        ln_kernel<<<ROWS_, 256, 0, stream>>>(x, b0ln2w, b0ln2b, xn, nullptr, 0, 1);
        run_dense_mlp(b0gw, b0gb, b0uw, b0ub, b0dw, b0db);
    }

    for (int r = 0; r < REC_; ++r) {
        run_attn(b1ln1w, b1ln1b, b1aw, b1ab, b1pw, b1pb);
        ln_kernel<<<ROWS_, 256, 0, stream>>>(x, b1ln2w, b1ln2b, xn, nullptr, 0, 1);
        router_kernel<<<ROWS_ / 8, 256, 0, stream>>>(
            xn, rw, rb, nw, nb, rnois + (size_t)r * B_ * T_ * E_, gate, topk);
        compact_kernel<<<E_, 256, 0, stream>>>(topk, lists, cnts);
        // shared expert: dense
        run_dense_mlp(sgw, sgb, suw, sub_, sdw, sdb);
        // routed experts: sparse gather/scatter dispatch (exact: dropped
        // experts have gate == 0.0 in the reference)
        for (int e = 0; e < E_; ++e) {
            const int* list = lists + (size_t)e * ROWS_;
            const int* cnt  = cnts + e;
            gemm_gather(stream, xn, egw + (size_t)e * HID_ * D_,
                        egb + (size_t)e * HID_, gbuf, list, cnt, HID_, D_);
            gemm_gather(stream, xn, euw + (size_t)e * HID_ * D_,
                        eub + (size_t)e * HID_, ubuf, list, cnt, HID_, D_);
            silu_mul_kernel<<<(ROWS_ * HID_) / 256, 256, 0, stream>>>(
                gbuf, ubuf, hbf, ROWS_ * HID_, cnt);
            gemm_scatter(stream, hbf, edw + (size_t)e * D_ * HID_,
                         edb + (size_t)e * D_, x, gate + e, E_, list, cnt, D_, HID_);
        }
    }

    ln_kernel<<<B_, 256, 0, stream>>>(x, lnfw, lnfb, nullptr, xnf, T_ - 1, T_);
    lmhead_kernel<<<V_, 256, 0, stream>>>(xnf, wte, (float*)d_out);
}
